// Model_69698729279519
// MI455X (gfx1250) — compile-verified, matrix operand order checked by reading, not run on hardware
//
#include <hip/hip_runtime.h>
#include <hip/hip_bf16.h>
#include <cstdint>
#include <cstddef>

#define BATCH_ 16
#define SEMB_  128
#define NCHA_  256

typedef __attribute__((ext_vector_type(16))) __bf16 v16bf;
typedef __attribute__((ext_vector_type(8)))  __bf16 v8bf;
typedef __attribute__((ext_vector_type(8)))  float  v8f;
typedef __attribute__((ext_vector_type(4)))  unsigned int v4u;
typedef __attribute__((ext_vector_type(8)))  int v8i;
typedef __attribute__((ext_vector_type(4)))  int v4i;

// Build a 16-element bf16 WMMA fragment from two contiguous 8-element runs in
// LDS (16-bit A/B striping: lane half g selects K ranges [8g,8g+7], [16+8g,16+8g+7]).
__device__ __forceinline__ v16bf frag_from_lds(const __bf16* p) {
    v8bf lo = *(const v8bf*)(p);
    v8bf hi = *(const v8bf*)(p + 16);
    return __builtin_shufflevector(lo, hi, 0,1,2,3,4,5,6,7,8,9,10,11,12,13,14,15);
}

#if __has_builtin(__builtin_amdgcn_tensor_load_to_lds)
#define USE_TDM 1
// Issue a 2D TDM tile load: tile (tile_d1 rows x tile_d0 elems, 2-byte elems)
// from a row-major tensor (row stride stride0 elems) into LDS at lds_off.
// D# bitfields per CDNA5 ISA ch.8 (group0: count/lds/global/type, group1: dims).
// clang-23 toolchain: 6-arg form (g0 v4u, g1 v8i, g2 v4i, g3 v4i, v8i, cpol).
__device__ __forceinline__ void tdm_load_2d_bf16(unsigned lds_off, const void* gptr,
                                                 unsigned tensor_d0, unsigned tensor_d1,
                                                 unsigned stride0,
                                                 unsigned tile_d0, unsigned tile_d1) {
    unsigned long long ga = (unsigned long long)(uintptr_t)gptr;
    v4u g0;
    g0[0] = 1u;                                        // count=1, user descriptor
    g0[1] = lds_off;                                   // LDS byte address
    g0[2] = (unsigned)(ga & 0xFFFFFFFFu);              // global_addr[31:0]
    g0[3] = (unsigned)((ga >> 32) & 0x1FFFFFFu)        // global_addr[56:32]
            | (2u << 30);                              // type=2 ("image")
    v8i g1;
    g1[0] = (int)(1u << 16);                           // data_size=1 -> 2 bytes
    g1[1] = (int)((tensor_d0 & 0xFFFFu) << 16);        // tensor_dim0[15:0]
    g1[2] = (int)((tensor_d0 >> 16) | ((tensor_d1 & 0xFFFFu) << 16));
    g1[3] = (int)((tensor_d1 >> 16) | (tile_d0 << 16));// tile_dim0
    g1[4] = (int)(tile_d1 & 0xFFFFu);                  // tile_dim1 (tile_dim2=0)
    g1[5] = (int)stride0;                              // tensor_dim0_stride[31:0]
    g1[6] = 0;                                         // stride0[47:32], stride1 lo
    g1[7] = 0;
    v4i z4 = {0, 0, 0, 0};
    v8i z8 = {0, 0, 0, 0, 0, 0, 0, 0};
    __builtin_amdgcn_tensor_load_to_lds(g0, g1, z4, z4, z8, 0);
}
#else
#define USE_TDM 0
#endif

// ---------------------------------------------------------------- init / util
__global__ void k_init(float* __restrict__ logdet) {
    if (threadIdx.x < BATCH_) logdet[threadIdx.x] = 0.0f;
}

__global__ __launch_bounds__(256) void k_cvt(const float* __restrict__ src,
                                             __bf16* __restrict__ dst, int n) {
    int i = blockIdx.x * 256 + threadIdx.x;
    if (i < n) dst[i] = (__bf16)src[i];
}

// squeeze: (B,nsq,Lc) -> (B,2nsq,Lc/2), new[b, 2c+q, l] = old[b, c, 2l+q]
__global__ __launch_bounds__(256) void k_squeeze(const float* __restrict__ src,
                                                 float* __restrict__ dst, int LcNew) {
    int idx = blockIdx.x * 256 + threadIdx.x;      // B*4096 total
    int b   = idx >> 12;
    int rem = idx & 4095;
    int cN  = rem / LcNew;
    int lN  = rem % LcNew;
    int c   = cN >> 1, q = cN & 1;
    int LcOld = LcNew * 2;
    dst[(size_t)b * 4096 + rem] =
        src[(size_t)b * 4096 + (size_t)c * LcOld + 2 * lN + q];
}

// ------------------------------------------------------- per-flow prep (tiny)
__global__ __launch_bounds__(256) void k_prep(const float* __restrict__ wl,
                                              const float* __restrict__ wu,
                                              const float* __restrict__ wsv,
                                              const float* __restrict__ logs,
                                              const float* __restrict__ wp,
                                              const float* __restrict__ ssign,
                                              float* __restrict__ Wout,
                                              float* __restrict__ logdet,
                                              int nsq, int Lc) {
    __shared__ float sL[256], sU[256], sT[256];
    __shared__ float sC;
    int tid = threadIdx.x;
    int n2  = nsq * nsq;
    if (tid < n2) {
        int i = tid / nsq, j = tid % nsq;
        sL[tid] = (i > j) ? wl[tid] : (i == j ? 1.0f : 0.0f);
        sU[tid] = (i < j) ? wu[tid] : (i == j ? ssign[i] * __expf(wsv[i]) : 0.0f);
    }
    if (tid == 0) {
        float c = 0.0f;
        for (int i = 0; i < nsq; ++i) c += wsv[i] + logs[i];
        sC = c * (float)Lc;
    }
    __syncthreads();
    if (tid < n2) {
        int i = tid / nsq, j = tid % nsq;
        float a = 0.0f;
        for (int k = 0; k < nsq; ++k) a = fmaf(sL[i * nsq + k], sU[k * nsq + j], a);
        sT[tid] = a;
    }
    __syncthreads();
    if (tid < n2) {
        int i = tid / nsq, j = tid % nsq;
        float a = 0.0f;
        for (int k = 0; k < nsq; ++k) a = fmaf(wp[i * nsq + k], sT[k * nsq + j], a);
        Wout[tid] = a;
    }
    if (tid < BATCH_) logdet[tid] += sC;
}

// ----------------------------------------- invertible 1x1 conv + affine (VALU)
__global__ __launch_bounds__(256) void k_affine(const float* __restrict__ src,
                                                float* __restrict__ dst,
                                                const float* __restrict__ W,
                                                const float* __restrict__ logs,
                                                const float* __restrict__ m,
                                                int nsq, int Lc) {
    int idx = blockIdx.x * 256 + threadIdx.x;     // B*Lc total
    int b = idx / Lc, l = idx % Lc;
    const float* sb = src + (size_t)b * 4096;
    float x[16];
    for (int j = 0; j < nsq; ++j) x[j] = sb[(size_t)j * Lc + l];
    float* db = dst + (size_t)b * 4096;
    for (int i = 0; i < nsq; ++i) {
        float a = 0.0f;
        for (int j = 0; j < nsq; ++j) a = fmaf(W[i * nsq + j], x[j], a);
        db[(size_t)i * Lc + l] = __expf(logs[i]) * (a + m[i]);
    }
}

// ----------------------------- embedding-conditioned weights: w(B,256,3), bb(B,256)
__global__ __launch_bounds__(256) void k_adapt(const float* __restrict__ emb_table,
                                               const int* __restrict__ s,
                                               const float* __restrict__ awW,
                                               const float* __restrict__ awb,
                                               const float* __restrict__ abW,
                                               const float* __restrict__ abb,
                                               float* __restrict__ wbuf,
                                               float* __restrict__ bbuf) {
    int idx = blockIdx.x * 256 + threadIdx.x;     // B*1024
    int b = idx >> 10, j = idx & 1023;
    const float* e = emb_table + (size_t)s[b] * SEMB_;
    if (j < 768) {
        const float* wr = awW + (size_t)j * SEMB_;
        float a = awb[j];
        for (int k = 0; k < SEMB_; ++k) a = fmaf(e[k], wr[k], a);
        wbuf[b * 768 + j] = a;
    } else {
        int jj = j - 768;
        const float* wr = abW + (size_t)jj * SEMB_;
        float a = abb[jj];
        for (int k = 0; k < SEMB_; ++k) a = fmaf(e[k], wr[k], a);
        bbuf[b * NCHA_ + jj] = a;
    }
}

// --------------- depthwise adapted conv + relu -> Y bf16, POSITION-MAJOR (b,l,ch)
// (channel contiguous so the GEMM B-slab is a plain 2D TDM tile, no transpose)
__global__ __launch_bounds__(256) void k_dw(const float* __restrict__ hh,
                                            const float* __restrict__ wbuf,
                                            const float* __restrict__ bbuf,
                                            __bf16* __restrict__ Y,
                                            int nh, int Lc) {
    long idx = (long)blockIdx.x * 256 + threadIdx.x;   // B*Lc*256
    int  ch  = (int)(idx & (NCHA_ - 1));
    long t   = idx >> 8;
    int  l   = (int)(t % Lc);
    int  b   = (int)(t / Lc);
    int  r   = NCHA_ / nh;
    int  c1  = ch / r;                                  // jnp.repeat semantics
    const float* hsrc = hh + (size_t)b * 4096 + (size_t)c1 * Lc;
    float acc = bbuf[b * NCHA_ + ch];
    const float* wp = wbuf + (size_t)b * 768 + ch * 3;
#pragma unroll
    for (int k = 0; k < 3; ++k) {
        int pos = l + k - 1;
        if (pos >= 0 && pos < Lc) acc = fmaf(wp[k], hsrc[pos], acc);
    }
    acc = acc > 0.0f ? acc : 0.0f;
    Y[idx] = (__bf16)acc;   // (b*Lc + l)*256 + ch
}

// ------------------------------ X = relu(c1_w @ Y + c1_b): bf16 WMMA, TDM staging
// WG = 8 waves computes a 256(M) x 64(N) slab for one batch row; K staged in
// 64-wide LDS slabs via tensor_load_to_lds (wave0: A tile, wave1: B tile).
__global__ __launch_bounds__(256) void k_gemm(const __bf16* __restrict__ c1bf,
                                              const float* __restrict__ c1b,
                                              const __bf16* __restrict__ Y,
                                              __bf16* __restrict__ X, int Lc) {
    __shared__ __bf16 sA[256 * 64];   // [m][k], k stride 1  (32 KB)
    __shared__ __bf16 sB[64 * 64];    // [n][k], k stride 1  ( 8 KB)
    const int b    = blockIdx.y;
    const int n0   = blockIdx.x * 64;
    const int tid  = threadIdx.x;
    const int wave = tid >> 5;
    const int lane = tid & 31;
    const int g    = lane >> 4;
    const int ln16 = lane & 15;
    v8f acc[2][4] = {};
    const __bf16* Yb = Y + ((size_t)b * Lc + n0) * NCHA_;  // row n -> 256 channels

    for (int k0 = 0; k0 < 256; k0 += 64) {
#if USE_TDM
        if (wave == 0) {
            tdm_load_2d_bf16((unsigned)(uintptr_t)&sA[0], c1bf + k0,
                             /*tensor_d0=*/256, /*tensor_d1=*/256,
                             /*stride0=*/256, /*tile_d0=*/64, /*tile_d1=*/256);
            __builtin_amdgcn_s_wait_tensorcnt(0);
        } else if (wave == 1) {
            tdm_load_2d_bf16((unsigned)(uintptr_t)&sB[0], Yb + k0,
                             /*tensor_d0=*/256, /*tensor_d1=*/64,
                             /*stride0=*/256, /*tile_d0=*/64, /*tile_d1=*/64);
            __builtin_amdgcn_s_wait_tensorcnt(0);
        }
#else
        for (int i = tid; i < 256 * 64; i += 256) {       // stage A (bf16 copy)
            int m = i >> 6, kk = i & 63;
            sA[i] = c1bf[m * 256 + k0 + kk];
        }
        for (int i = tid; i < 64 * 64; i += 256) {        // stage B ([n][k])
            int n = i >> 6, kk = i & 63;
            sB[i] = Yb[(size_t)n * NCHA_ + k0 + kk];
        }
#endif
        __syncthreads();
#pragma unroll
        for (int ks = 0; ks < 2; ++ks) {                  // two K=32 steps per slab
            v16bf af[2], bfr[4];
#pragma unroll
            for (int t = 0; t < 2; ++t) {
                int row = (wave * 2 + t) * 16 + ln16;
                af[t] = frag_from_lds(&sA[row * 64 + ks * 32 + 8 * g]);
            }
#pragma unroll
            for (int t = 0; t < 4; ++t) {
                int col = t * 16 + ln16;
                bfr[t] = frag_from_lds(&sB[col * 64 + ks * 32 + 8 * g]);
            }
#pragma unroll
            for (int t2 = 0; t2 < 2; ++t2)
#pragma unroll
                for (int t = 0; t < 4; ++t)
                    acc[t2][t] = __builtin_amdgcn_wmma_f32_16x16x32_bf16(
                        false, af[t2], false, bfr[t], (short)0, acc[t2][t],
                        false, false);
        }
        __syncthreads();
    }
    // epilogue: bias + relu; D layout: VGPR r -> row = tile*16 + r + 8*g, col = n16
#pragma unroll
    for (int t2 = 0; t2 < 2; ++t2)
#pragma unroll
        for (int t = 0; t < 4; ++t)
#pragma unroll
            for (int r = 0; r < 8; ++r) {
                int row = (wave * 2 + t2) * 16 + r + 8 * g;
                int col = n0 + t * 16 + ln16;
                float v = acc[t2][t][r] + c1b[row];
                v = v > 0.0f ? v : 0.0f;
                X[(size_t)b * NCHA_ * Lc + (size_t)row * Lc + col] = (__bf16)v;
            }
}

// ----------- output conv (2nh,256,3) + gate + h2 update + log-det accumulation
__global__ __launch_bounds__(256) void k_post(const __bf16* __restrict__ X,
                                              const float* __restrict__ c2w,
                                              const float* __restrict__ c2b,
                                              float* __restrict__ hh,
                                              float* __restrict__ logdet,
                                              int nh, int Lc) {
    int idx = blockIdx.x * 256 + threadIdx.x;   // B*nh*Lc == 32768 always
    int b   = idx / (nh * Lc);
    int rem = idx % (nh * Lc);
    int c   = rem / Lc;
    int l   = rem % Lc;
    float xs = c2b[c], xm = c2b[c + nh];
    const float*  ws_ = c2w + (size_t)c * NCHA_ * 3;
    const float*  wm_ = c2w + (size_t)(c + nh) * NCHA_ * 3;
    const __bf16* Xb  = X + (size_t)b * NCHA_ * Lc;
#pragma unroll
    for (int k = 0; k < 3; ++k) {
        int pos = l + k - 1;
        if (pos < 0 || pos >= Lc) continue;
        for (int ic = 0; ic < NCHA_; ++ic) {
            float xv = (float)Xb[(size_t)ic * Lc + pos];
            xs = fmaf(ws_[ic * 3 + k], xv, xs);
            xm = fmaf(wm_[ic * 3 + k], xv, xm);
        }
    }
    float sg = 1.0f / (1.0f + __expf(-(xs + 2.0f))) + 1e-7f;
    float* hp = hh + (size_t)b * 4096 + (size_t)(nh + c) * Lc + l;
    *hp = sg * (*hp + xm);
    float ld = __logf(sg);
    for (int o = 16; o > 0; o >>= 1) ld += __shfl_down(ld, o);   // wave32 reduce
    if ((threadIdx.x & 31) == 0) atomicAdd(&logdet[b], ld);
}

__global__ void k_final(const float* __restrict__ hh,
                        const float* __restrict__ logdet,
                        float* __restrict__ out) {
    int idx = blockIdx.x * 256 + threadIdx.x;
    if (idx < 65536)      out[idx] = hh[idx];
    else if (idx < 65552) out[idx] = logdet[idx - 65536];
}

// ---------------------------------------------------------------------- host
extern "C" void kernel_launch(void* const* d_in, const int* in_sizes, int n_in,
                              void* d_out, int out_size, void* d_ws, size_t ws_size,
                              hipStream_t stream) {
    (void)in_sizes; (void)n_in; (void)out_size; (void)ws_size;
    const float* h   = (const float*)d_in[0];
    const int*   s   = (const int*)d_in[1];
    const float* emb = (const float*)d_in[2];
    // params dict insertion order:
    // 0 w_l 1 w_u 2 w_s 3 m 4 logs 5 aw_W 6 aw_b 7 ab_W 8 ab_b 9 c1_w 10 c1_b 11 c2_w 12 c2_b
    auto P   = [&](int blk, int fl, int slot) {
        return (const float*)d_in[3 + (blk * 8 + fl) * 13 + slot];
    };
    auto BUF = [&](int blk, int fl, int slot) {   // 0 w_p, 1 s_sign
        return (const float*)d_in[3 + 4 * 8 * 13 + (blk * 8 + fl) * 2 + slot];
    };

    char*  w   = (char*)d_ws;
    size_t off = 0;
    auto carve = [&](size_t bytes) -> char* {
        char* p = w + off;
        off += (bytes + 255) & ~(size_t)255;
        return p;
    };
    float*  hhA    = (float*)carve((size_t)BATCH_ * 4096 * 4);
    float*  hhB    = (float*)carve((size_t)BATCH_ * 4096 * 4);
    float*  logdet = (float*)carve(64);
    float*  Wmat   = (float*)carve(16 * 16 * 4);
    float*  wbuf   = (float*)carve((size_t)BATCH_ * 768 * 4);
    float*  bbuf   = (float*)carve((size_t)BATCH_ * 256 * 4);
    __bf16* c1bf   = (__bf16*)carve((size_t)NCHA_ * NCHA_ * 2);
    __bf16* Ybf    = (__bf16*)carve((size_t)BATCH_ * NCHA_ * 2048 * 2);
    __bf16* Xbf    = (__bf16*)carve((size_t)BATCH_ * NCHA_ * 2048 * 2);

    k_init<<<1, 32, 0, stream>>>(logdet);

    int nsq = 1, Lc = 4096;
    for (int blk = 0; blk < 4; ++blk) {
        k_squeeze<<<256, 256, 0, stream>>>(blk == 0 ? h : hhA, hhB, Lc / 2);
        { float* t = hhA; hhA = hhB; hhB = t; }
        nsq *= 2; Lc /= 2;
        int nh = nsq / 2;
        for (int fl = 0; fl < 8; ++fl) {
            k_prep<<<1, 256, 0, stream>>>(P(blk, fl, 0), P(blk, fl, 1),
                                          P(blk, fl, 2), P(blk, fl, 4),
                                          BUF(blk, fl, 0), BUF(blk, fl, 1),
                                          Wmat, logdet, nsq, Lc);
            k_affine<<<(BATCH_ * Lc) / 256, 256, 0, stream>>>(
                hhA, hhB, Wmat, P(blk, fl, 4), P(blk, fl, 3), nsq, Lc);
            k_adapt<<<64, 256, 0, stream>>>(emb, s, P(blk, fl, 5), P(blk, fl, 6),
                                            P(blk, fl, 7), P(blk, fl, 8),
                                            wbuf, bbuf);
            k_cvt<<<(NCHA_ * NCHA_) / 256, 256, 0, stream>>>(P(blk, fl, 9), c1bf,
                                                             NCHA_ * NCHA_);
            k_dw<<<(BATCH_ * NCHA_ * Lc) / 256, 256, 0, stream>>>(
                hhB, wbuf, bbuf, Ybf, nh, Lc);
            dim3 g5(Lc / 64, BATCH_);
            k_gemm<<<g5, 256, 0, stream>>>(c1bf, P(blk, fl, 10), Ybf, Xbf, Lc);
            k_post<<<128, 256, 0, stream>>>(Xbf, P(blk, fl, 11), P(blk, fl, 12),
                                            hhB, logdet, nh, Lc);
            { float* t = hhA; hhA = hhB; hhB = t; }
        }
    }
    k_final<<<257, 256, 0, stream>>>(hhA, logdet, (float*)d_out);
}
